// PointnetSAMoudleAgg_5085241279181
// MI455X (gfx1250) — matrix operand order repaired and checked by reading, not compiled, not run on hardware
//
#include <hip/hip_runtime.h>
#include <hip/hip_bf16.h>

// ---------------- problem constants ----------------
#define BQ_B     4
#define BQ_N     16384
#define BQ_P     1024
#define NSAMP    64
#define CFEAT    128
#define MTOT     262144      // B*P*S
#define MT       16384       // MTOT/16 (M tiles)
#define RADIUS_F 0.4f
#define BN_EPS_F 1e-5f

typedef __attribute__((ext_vector_type(16))) __bf16 v16bf;
typedef __attribute__((ext_vector_type(8)))  float  v8f;

// float -> bf16 round-to-nearest-even
__device__ __forceinline__ unsigned short f2bf(float f) {
    unsigned u = __float_as_uint(f);
    unsigned r = u + 0x7FFFu + ((u >> 16) & 1u);
    return (unsigned short)(r >> 16);
}

// load one 32-byte WMMA fragment slice for this lane (two b128 loads)
__device__ __forceinline__ v16bf load_frag(const unsigned short* p) {
    union { int4 i[2]; v16bf v; } u;
    const int4* q = (const int4*)p;
    u.i[0] = q[0];
    u.i[1] = q[1];
    return u.v;
}

// ---------------- 1) ball query: one wave32 per query point ----------------
__global__ __launch_bounds__(256) void ball_query_kernel(
        const float* __restrict__ xyz, const float* __restrict__ nxyz,
        int* __restrict__ idx) {
    int gw   = (blockIdx.x * 256 + threadIdx.x) >> 5;   // global wave = query id
    int lane = threadIdx.x & 31;
    int b = gw >> 10, p = gw & 1023;
    const float* q = nxyz + (size_t)(b * BQ_P + p) * 3;
    float qx = q[0], qy = q[1], qz = q[2];
    const float* xb = xyz + (size_t)b * BQ_N * 3;
    int* out = idx + (size_t)gw * NSAMP;
    const float r2 = RADIUS_F * RADIUS_F;
    int cnt = 0;
    for (int n0 = 0; n0 < BQ_N && cnt < NSAMP; n0 += 32) {
        int n = n0 + lane;
        const float* pt = xb + n * 3;
        float dx = pt[0] - qx, dy = pt[1] - qy, dz = pt[2] - qz;
        bool pred = (dx * dx + dy * dy + dz * dz) < r2;
        unsigned m32 = __builtin_amdgcn_ballot_w32(pred);
        if (pred) {
            int slot = cnt + __popc(m32 & ((1u << lane) - 1u));
            if (slot < NSAMP) out[slot] = n;
        }
        cnt += __popc(m32);
    }
    cnt = min(cnt, NSAMP);
    int first = (cnt > 0) ? out[0] : 0;   // same-wave store->load same addr is ordered
    for (int s = cnt + lane; s < NSAMP; s += 32) out[s] = first;
}

// ---------------- 2) pack W into A-fragment layout (16x32 bf16 tiles) ------
// A tile (M=16,K=32): lane = hi*16 + row(M); element e (0..15) in v16h:
//   klocal = hi*16 + e
__global__ __launch_bounds__(256) void pack_W_kernel(
        const float* __restrict__ W, unsigned short* __restrict__ Wp,
        int Cin, int KS, int total) {
    int gid = blockIdx.x * 256 + threadIdx.x;
    if (gid >= total) return;
    int t = gid >> 9, r = gid & 511;
    int lane = r >> 4, e = r & 15;
    int ot = t / KS, ks = t - ot * KS;
    int hi = lane >> 4, row = lane & 15;
    int klocal = hi * 16 + e;
    int k = ks * 32 + klocal;
    int o = ot * 16 + row;
    float v = (k < Cin) ? W[o * Cin + k] : 0.f;
    Wp[gid] = f2bf(v);
}

// ---------------- 3) pack X0 (gather+concat) into B-fragment layout --------
// B tile (K=32,N=16): lane = hi*16 + col(N); element e (0..15): klocal = hi*16 + e
__global__ __launch_bounds__(256) void pack_X0_kernel(
        const float* __restrict__ xyz, const float* __restrict__ nxyz,
        const float* __restrict__ feat, const int* __restrict__ idx,
        unsigned short* __restrict__ Xa) {
    int gid = blockIdx.x * 256 + threadIdx.x;        // total = 5*MT*512
    int t = gid >> 9, r = gid & 511;
    int lane = r >> 4, e = r & 15;
    int ks = t / MT, mt = t - ks * MT;
    int hi = lane >> 4, col = lane & 15;
    int k = ks * 32 + hi * 16 + e;
    int m = (mt << 4) + col;
    int b = m >> 16; int rem = m & 65535;
    int p = rem >> 6; int s = rem & 63;
    int id = idx[((b << 10) + p) * NSAMP + s];
    float val;
    if (k < 3) {
        val = (xyz[((size_t)b * BQ_N + id) * 3 + k] -
               nxyz[((size_t)(b * BQ_P + p)) * 3 + k]) * (1.0f / RADIUS_F);
    } else if (k < 131) {
        val = feat[((size_t)(b * CFEAT + (k - 3))) * BQ_N + id];
    } else {
        val = 0.f;
    }
    Xa[gid] = f2bf(val);
}

// ---------------- 4) bf16 WMMA GEMM + fused BN statistics ------------------
// Wave computes a 16(O) x 64(M) strip = 4 C tiles, software-pipelined K loop.
// Epilogue: per-channel sum/sum^2 reduced via shuffles + LDS + global atomics
// (all 8 waves of a block share the same ot => same 16 channels).
template <int KS>
__global__ __launch_bounds__(256) void gemm_bf16_kernel(
        const unsigned short* __restrict__ Ap,
        const unsigned short* __restrict__ Bp,
        float* __restrict__ Yp,
        float* __restrict__ sum, float* __restrict__ sumsq) {
    int tid  = threadIdx.x;
    int wave = (blockIdx.x * 256 + tid) >> 5;
    int lane = tid & 31;
    int ot  = wave >> 12;               // wave / (MT/4)
    int mt0 = (wave & 4095) << 2;       // 4 M-tiles per wave

    __shared__ float bsum[16], bsq[16];
    if (tid < 16) { bsum[tid] = 0.f; bsq[tid] = 0.f; }
    __syncthreads();

    v8f acc[4] = {};
    const unsigned short* ap = Ap + (size_t)ot * KS * 512 + lane * 16;
    const unsigned short* bp = Bp + (((size_t)mt0) << 9) + lane * 16;
    const size_t bstep = (size_t)MT << 9;   // elements per K-step in packed B

    v16bf a_c  = load_frag(ap);
    v16bf b_c0 = load_frag(bp);
    v16bf b_c1 = load_frag(bp + 512);
    v16bf b_c2 = load_frag(bp + 1024);
    v16bf b_c3 = load_frag(bp + 1536);

#pragma unroll
    for (int ks = 0; ks + 1 < KS; ++ks) {
        const unsigned short* bpn = bp + (size_t)(ks + 1) * bstep;
        v16bf a_n  = load_frag(ap + (ks + 1) * 512);
        v16bf b_n0 = load_frag(bpn);
        v16bf b_n1 = load_frag(bpn + 512);
        v16bf b_n2 = load_frag(bpn + 1024);
        v16bf b_n3 = load_frag(bpn + 1536);
        acc[0] = __builtin_amdgcn_wmma_f32_16x16x32_bf16(
            false, a_c, false, b_c0, (short)0, acc[0], false, false);
        acc[1] = __builtin_amdgcn_wmma_f32_16x16x32_bf16(
            false, a_c, false, b_c1, (short)0, acc[1], false, false);
        acc[2] = __builtin_amdgcn_wmma_f32_16x16x32_bf16(
            false, a_c, false, b_c2, (short)0, acc[2], false, false);
        acc[3] = __builtin_amdgcn_wmma_f32_16x16x32_bf16(
            false, a_c, false, b_c3, (short)0, acc[3], false, false);
        a_c = a_n; b_c0 = b_n0; b_c1 = b_n1; b_c2 = b_n2; b_c3 = b_n3;
    }
    // epilogue: last K-step
    acc[0] = __builtin_amdgcn_wmma_f32_16x16x32_bf16(
        false, a_c, false, b_c0, (short)0, acc[0], false, false);
    acc[1] = __builtin_amdgcn_wmma_f32_16x16x32_bf16(
        false, a_c, false, b_c1, (short)0, acc[1], false, false);
    acc[2] = __builtin_amdgcn_wmma_f32_16x16x32_bf16(
        false, a_c, false, b_c2, (short)0, acc[2], false, false);
    acc[3] = __builtin_amdgcn_wmma_f32_16x16x32_bf16(
        false, a_c, false, b_c3, (short)0, acc[3], false, false);

    // store C tiles (packed layout)
#pragma unroll
    for (int i = 0; i < 4; ++i) {
        float* dst = Yp + (((size_t)(ot * MT + mt0 + i)) << 8) + lane * 8;
        float4 lo = { acc[i][0], acc[i][1], acc[i][2], acc[i][3] };
        float4 hi = { acc[i][4], acc[i][5], acc[i][6], acc[i][7] };
        ((float4*)dst)[0] = lo;
        ((float4*)dst)[1] = hi;
    }

    // fused BN stats: lane-local accumulation over its column x 4 tiles
    float s8[8], q8[8];
#pragma unroll
    for (int v = 0; v < 8; ++v) { s8[v] = 0.f; q8[v] = 0.f; }
#pragma unroll
    for (int i = 0; i < 4; ++i) {
#pragma unroll
        for (int v = 0; v < 8; ++v) {
            float x = acc[i][v];
            s8[v] += x; q8[v] += x * x;
        }
    }
    // reduce across the 16 columns (xor masks < 16 stay within half-wave)
#pragma unroll
    for (int m = 1; m <= 8; m <<= 1) {
#pragma unroll
        for (int v = 0; v < 8; ++v) {
            s8[v] += __shfl_xor(s8[v], m);
            q8[v] += __shfl_xor(q8[v], m);
        }
    }
    if ((lane & 15) == 0) {
        int hi = lane >> 4;               // channel group: o_local = hi*8 + v
#pragma unroll
        for (int v = 0; v < 8; ++v) {
            atomicAdd(&bsum[hi * 8 + v], s8[v]);
            atomicAdd(&bsq[hi * 8 + v], q8[v]);
        }
    }
    __syncthreads();
    if (tid < 16) {
        atomicAdd(&sum[ot * 16 + tid], bsum[tid]);
        atomicAdd(&sumsq[ot * 16 + tid], bsq[tid]);
    }
}

__global__ void zero_kernel(float* p, int n) {
    int i = blockIdx.x * 256 + threadIdx.x;
    if (i < n) p[i] = 0.f;
}

__global__ void scale_shift_kernel(const float* __restrict__ sum,
                                   const float* __restrict__ sumsq,
                                   const float* __restrict__ g,
                                   const float* __restrict__ be,
                                   float* __restrict__ scale,
                                   float* __restrict__ shift, int O) {
    int o = threadIdx.x;
    if (o < O) {
        const float inv = 1.0f / (float)MTOT;
        float mean = sum[o] * inv;
        float var  = sumsq[o] * inv - mean * mean;
        float sc   = g[o] * rsqrtf(var + BN_EPS_F);
        scale[o] = sc;
        shift[o] = be[o] - mean * sc;
    }
}

// ---------------- 5) BN+ReLU, repack as next layer's B-fragments (bf16) ----
__global__ __launch_bounds__(256) void normalize_pack_kernel(
        const float* __restrict__ Yp, const float* __restrict__ scale,
        const float* __restrict__ shift, unsigned short* __restrict__ Xn) {
    int gid = blockIdx.x * 256 + threadIdx.x;      // total = O*MTOT
    int t = gid >> 8, r = gid & 255;
    int ot = t / MT, mt = t - ot * MT;
    int lane = r >> 3, v = r & 7;
    int hi = lane >> 4, col = lane & 15;
    int o = ot * 16 + hi * 8 + v;
    float y = Yp[gid];
    float val = fmaxf(0.f, y * scale[o] + shift[o]);
    int ks2 = o >> 5, kl = o & 31, hb = kl >> 4, eb = kl & 15;
    size_t addr = (((size_t)(ks2 * MT + mt)) << 9) + ((hb * 16 + col) << 4) + eb;
    Xn[addr] = f2bf(val);
}

// ---------------- 6) final fused BN+ReLU+max over S ------------------------
__global__ __launch_bounds__(256) void final_max_kernel(
        const float* __restrict__ Yp, const float* __restrict__ scale,
        const float* __restrict__ shift, float* __restrict__ out) {
    int gid = blockIdx.x * 256 + threadIdx.x;   // B*256*1024 threads, (b,o,p)
    int p = gid & 1023;
    int t2 = gid >> 10;
    int o = t2 & 255;
    int b = t2 >> 8;
    int ot = o >> 4, ol = o & 15;
    int hi = ol >> 3, v = ol & 7;
    float sc = scale[o], sh = shift[o];
    int m_base = (b * BQ_P + p) * NSAMP;
    float best = 0.f;                           // relu => >= 0
    for (int s = 0; s < NSAMP; ++s) {
        int m = m_base + s;
        int mt = m >> 4, col = m & 15;
        int lane = hi * 16 + col;
        size_t a = (((size_t)(ot * MT + mt)) << 8) + lane * 8 + v;
        float y = Yp[a];
        best = fmaxf(best, fmaxf(0.f, y * sc + sh));
    }
    out[gid] = best;
}

// ---------------- workspace offsets (bytes) --------------------------------
#define OFF_IDX   ((size_t)0)                       // 262144*4   = 1 MB
#define OFF_W0P   ((size_t)1048576)                 // 20480*2
#define OFF_W1P   ((size_t)(OFF_W0P + 40960))       // 16384*2
#define OFF_W2P   ((size_t)(OFF_W1P + 32768))       // 32768*2
#define OFF_STATS ((size_t)(OFF_W2P + 65536))       // 1024 floats
#define OFF_XA    ((size_t)(OFF_STATS + 4352))      // 5*MT*512*2 = 80 MB
#define OFF_XB    ((size_t)(OFF_XA + (size_t)5*MT*512*2))  // 4*MT*512*2 = 64 MB
#define OFF_Y     ((size_t)(OFF_XB + (size_t)4*MT*512*2))  // 16*MT*256*4 = 256 MB

extern "C" void kernel_launch(void* const* d_in, const int* in_sizes, int n_in,
                              void* d_out, int out_size, void* d_ws, size_t ws_size,
                              hipStream_t stream) {
    const float* xyz  = (const float*)d_in[0];
    const float* nxyz = (const float*)d_in[1];
    const float* feat = (const float*)d_in[2];
    const float* W0 = (const float*)d_in[3];
    const float* g0 = (const float*)d_in[4];
    const float* b0 = (const float*)d_in[5];
    const float* W1 = (const float*)d_in[6];
    const float* g1 = (const float*)d_in[7];
    const float* b1 = (const float*)d_in[8];
    const float* W2 = (const float*)d_in[9];
    const float* g2 = (const float*)d_in[10];
    const float* b2 = (const float*)d_in[11];

    char* ws = (char*)d_ws;
    int*            idx  = (int*)(ws + OFF_IDX);
    unsigned short* W0p  = (unsigned short*)(ws + OFF_W0P);
    unsigned short* W1p  = (unsigned short*)(ws + OFF_W1P);
    unsigned short* W2p  = (unsigned short*)(ws + OFF_W2P);
    float*          stats = (float*)(ws + OFF_STATS);
    float* sum = stats, *sumsq = stats + 256, *scl = stats + 512, *sft = stats + 768;
    unsigned short* Xa = (unsigned short*)(ws + OFF_XA);
    unsigned short* Xb = (unsigned short*)(ws + OFF_XB);
    float*          Yp = (float*)(ws + OFF_Y);
    float*          out = (float*)d_out;

    // 1) ball query: 4096 waves = 512 blocks
    ball_query_kernel<<<512, 256, 0, stream>>>(xyz, nxyz, idx);

    // 2) pack weights (K padded to 160 for layer 0)
    pack_W_kernel<<<80, 256, 0, stream>>>(W0, W0p, 131, 5, 8 * 5 * 512);
    pack_W_kernel<<<64, 256, 0, stream>>>(W1, W1p, 128, 4, 8 * 4 * 512);
    pack_W_kernel<<<128, 256, 0, stream>>>(W2, W2p, 128, 4, 16 * 4 * 512);

    // 3) gather + concat + pack X0 (5*MT*512 = 41,943,040 elements)
    pack_X0_kernel<<<163840, 256, 0, stream>>>(xyz, nxyz, feat, idx, Xa);

    // ---- layer 0: 131(pad160) -> 128 ----
    zero_kernel<<<2, 256, 0, stream>>>(stats, 512);
    gemm_bf16_kernel<5><<<4096, 256, 0, stream>>>(W0p, Xa, Yp, sum, sumsq);
    scale_shift_kernel<<<1, 256, 0, stream>>>(sum, sumsq, g0, b0, scl, sft, 128);
    normalize_pack_kernel<<<131072, 256, 0, stream>>>(Yp, scl, sft, Xb);

    // ---- layer 1: 128 -> 128 ----
    zero_kernel<<<2, 256, 0, stream>>>(stats, 512);
    gemm_bf16_kernel<4><<<4096, 256, 0, stream>>>(W1p, Xb, Yp, sum, sumsq);
    scale_shift_kernel<<<1, 256, 0, stream>>>(sum, sumsq, g1, b1, scl, sft, 128);
    normalize_pack_kernel<<<131072, 256, 0, stream>>>(Yp, scl, sft, Xa);

    // ---- layer 2: 128 -> 256, fused BN+ReLU+max(S) ----
    zero_kernel<<<2, 256, 0, stream>>>(stats, 512);
    gemm_bf16_kernel<4><<<8192, 256, 0, stream>>>(W2p, Xa, Yp, sum, sumsq);
    scale_shift_kernel<<<1, 256, 0, stream>>>(sum, sumsq, g2, b2, scl, sft, 256);
    final_max_kernel<<<4096, 256, 0, stream>>>(Yp, scl, sft, out);
}
